// RPN_85598698209576
// MI455X (gfx1250) — matrix operand order repaired and checked by reading, not compiled, not run on hardware
//
#include <hip/hip_runtime.h>
#include <hip/hip_bf16.h>

typedef __attribute__((ext_vector_type(16))) _Float16 v16h;
typedef __attribute__((ext_vector_type(8)))  float    v8f;
typedef __attribute__((ext_vector_type(4)))  unsigned int u32x4;
typedef __attribute__((ext_vector_type(8)))  int      i32x8;
typedef __attribute__((ext_vector_type(4)))  int      i32x4;

union V16 { uint4 u[2]; v16h h; };

#if __has_builtin(__builtin_amdgcn_tensor_load_to_lds) && \
    __has_builtin(__builtin_amdgcn_s_wait_tensorcnt)
#define USE_TDM 1
#endif

// ---------------------------------------------------------------------------
// Pack base_w [3][3][512][256] f32 -> hi/lo f16 planes in B-fragment layout.
// Layout (halves): block per (tap,q): [plane(hi=0,lo=1)][n(0..255)][k(0..31)]
//   offset = ((tap*16+q)*2 + plane)*8192 + n*32 + k
// A lane of the WMMA B-fragment then reads 2x16B contiguous per n row.
// ---------------------------------------------------------------------------
__global__ __launch_bounds__(256) void rpn_pack_w(const float* __restrict__ bw,
                                                  _Float16* __restrict__ wp) {
    int idx = blockIdx.x * 256 + threadIdx.x;   // < 9*512*256 = 1179648
    float v = bw[idx];
    int n   = idx & 255;
    int c   = (idx >> 8) & 511;
    int tap = idx >> 17;
    _Float16 hi = (_Float16)v;
    _Float16 lo = (_Float16)(v - (float)hi);
    int q = c >> 5, kl = c & 31;
    size_t base = ((size_t)(tap * 16 + q) * 2) * 8192 + (size_t)n * 32 + kl;
    wp[base]        = hi;   // hi plane
    wp[base + 8192] = lo;   // lo plane
}

// ---------------------------------------------------------------------------
// Fused RPN kernel: 3x3 conv (implicit GEMM, f16x3 WMMA, f32 accum) + ReLU
// + 1x1 cls/reg heads + sigmoid + per-anchor box filtering, dense [B,H,W,45].
// Block: 512 threads = 16 waves, tile M=128 positions (2 rows) x N=256.
// Wave grid 8(M) x 2(N): each wave 16x128 -> 8 n-tiles x v8f accumulators.
// Weight chunks staged global->LDS by the Tensor Data Mover (one 2D
// descriptor per (tap,q): 16 dwords x 512 rows, LDS-padded 16B per 64B row
// for bank-conflict-free ds_load_b128 fragment reads).
// ---------------------------------------------------------------------------
#define LDS_ROWB   80                  // padded bytes per staged n-row (64B data)
#define LDS_PLANE  (256 * LDS_ROWB)    // 20480 B per hi/lo plane

__global__ __launch_bounds__(512) void rpn_main(
    const float* __restrict__ x,       // [32][64][64][512]
    const float* __restrict__ base_b,  // [256]
    const float* __restrict__ cls_w,   // [256][9]
    const float* __restrict__ cls_b,   // [9]
    const float* __restrict__ reg_w,   // [256][36]
    const float* __restrict__ reg_b,   // [36]
    const _Float16* __restrict__ wp,   // packed base weights (hi/lo)
    float* __restrict__ out)           // [32][64][64][45]
{
    __shared__ unsigned char smem[41984] __attribute__((aligned(16)));

    const int tid    = threadIdx.x;
    const int lane   = tid & 31;
    const int wave   = tid >> 5;
    const int mi     = wave >> 1;          // 0..7  (M wave)
    const int ni     = wave & 1;           // 0..1  (N wave)
    const int lane15 = lane & 15;
    const int ksel   = lane >> 4;          // which K half of the fragment
    const int bIdx   = blockIdx.x >> 5;    // batch 0..31
    const int h0     = (blockIdx.x & 31) * 2;
    const int mrow   = mi * 16 + lane15;   // 0..127 within tile
    const int hloc   = mrow >> 6;
    const int wloc   = mrow & 63;

    v8f acc[8];
    v8f vzero = {0.f, 0.f, 0.f, 0.f, 0.f, 0.f, 0.f, 0.f};
#pragma unroll
    for (int i = 0; i < 8; ++i) acc[i] = vzero;

    for (int tap = 0; tap < 9; ++tap) {
        const int dy = tap / 3 - 1, dx = tap % 3 - 1;
        const int hs = h0 + hloc + dy;
        const int ws = wloc + dx;
        const bool valid = (hs >= 0) & (hs < 64) & (ws >= 0) & (ws < 64);
        const int hsc = hs < 0 ? 0 : (hs > 63 ? 63 : hs);
        const int wsc = ws < 0 ? 0 : (ws > 63 ? 63 : ws);
        const float vm = valid ? 1.0f : 0.0f;
        const float* ap = x + ((size_t)((bIdx * 64 + hsc) * 64 + wsc)) * 512 + ksel * 16;

        for (int q = 0; q < 16; ++q) {
            // ---- stage one (tap,q) weight chunk (hi+lo) into LDS -------
#ifdef USE_TDM
            if (wave == 0) {
                // Tensor DMA: 2D tile, 16 dwords/row x 512 rows, contiguous in
                // global, 4-dword pad after every 16 dwords in LDS (64B->80B).
                unsigned long long ga =
                    (unsigned long long)(const void*)(wp + (size_t)(tap * 16 + q) * 16384);
                unsigned int lbase = (unsigned int)(unsigned long long)(uintptr_t)smem;
                u32x4 g0;
                g0[0] = 1u;                                  // count=1 (valid user D#)
                g0[1] = lbase;                               // lds_addr
                g0[2] = (unsigned int)ga;                    // global_addr[31:0]
                g0[3] = (unsigned int)(ga >> 32) | (2u << 30); // [56:32] | type=2
                i32x8 g1;
                g1[0] = (2 << 16)        // data_size = 4B
                      | (1 << 20)        // pad_enable
                      | (3 << 22)        // pad_interval: 16 dwords
                      | (3 << 25);       // pad_amount:   4 dwords
                g1[1] = (16 << 16);      // tensor_dim0 = 16
                g1[2] = (512 << 16);     // tensor_dim1 = 512
                g1[3] = (16 << 16);      // tile_dim0 = 16
                g1[4] = 512;             // tile_dim1 = 512
                g1[5] = 16;              // tensor_dim0_stride = 16
                g1[6] = 0;
                g1[7] = 0;
                i32x4 gz4 = {0, 0, 0, 0};
                i32x8 gz8 = {0, 0, 0, 0, 0, 0, 0, 0};
                __builtin_amdgcn_tensor_load_to_lds(g0, g1, gz4, gz4, gz8, 0);
                __builtin_amdgcn_s_wait_tensorcnt(0);
            }
#else
            {
                const int plane = tid >> 8;       // 0..1
                const int n     = tid & 255;
                const uint4* s4 =
                    (const uint4*)(wp + ((size_t)(tap * 16 + q) * 2 + plane) * 8192 +
                                   (size_t)n * 32);
                uint4* d4 = (uint4*)(smem + plane * LDS_PLANE + n * LDS_ROWB);
                d4[0] = s4[0]; d4[1] = s4[1]; d4[2] = s4[2]; d4[3] = s4[3];
            }
#endif
            __syncthreads();

            // ---- A fragment: 16 f32 per lane, split hi/lo -------------
            const float* a0 = ap + q * 32;
            if (q < 15) __builtin_prefetch(a0 + 32, 0, 1);   // global_prefetch_b8
            float4 f0 = *(const float4*)(a0);
            float4 f1 = *(const float4*)(a0 + 4);
            float4 f2 = *(const float4*)(a0 + 8);
            float4 f3 = *(const float4*)(a0 + 12);
            float fs[16] = {f0.x, f0.y, f0.z, f0.w, f1.x, f1.y, f1.z, f1.w,
                            f2.x, f2.y, f2.z, f2.w, f3.x, f3.y, f3.z, f3.w};
            v16h ahi, alo;
#pragma unroll
            for (int e = 0; e < 16; ++e) {
                float f = fs[e] * vm;
                _Float16 h = (_Float16)f;
                ahi[e] = h;
                alo[e] = (_Float16)(f - (float)h);
            }

            // ---- 8 n-tiles x (hi*hi + hi*lo + lo*hi) ------------------
#pragma unroll
            for (int nt = 0; nt < 8; ++nt) {
                const int n = (ni * 8 + nt) * 16 + lane15;
                const unsigned char* bp = smem + n * LDS_ROWB + ksel * 32;
                V16 bh, bl;
                bh.u[0] = *(const uint4*)(bp);
                bh.u[1] = *(const uint4*)(bp + 16);
                bl.u[0] = *(const uint4*)(bp + LDS_PLANE);
                bl.u[1] = *(const uint4*)(bp + LDS_PLANE + 16);
                acc[nt] = __builtin_amdgcn_wmma_f32_16x16x32_f16(
                    false, ahi, false, bh.h, (short)0, acc[nt], false, false);
                acc[nt] = __builtin_amdgcn_wmma_f32_16x16x32_f16(
                    false, ahi, false, bl.h, (short)0, acc[nt], false, false);
                acc[nt] = __builtin_amdgcn_wmma_f32_16x16x32_f16(
                    false, alo, false, bh.h, (short)0, acc[nt], false, false);
            }
            __syncthreads();
        }
    }

    // ---------------- fused epilogue, 32-row chunks ------------------------
    float* feat = (float*)smem;            // [32][260] f32 (padded rows)
    const int m_ep = tid >> 4;             // 0..31
    const int j    = tid & 15;             // anchor id when < 9

    for (int chunk = 0; chunk < 4; ++chunk) {
        __syncthreads();
        if ((mi >> 1) == chunk) {          // waves owning these 32 rows
            const int lmb = (mi & 1) * 16 + ksel * 8;
#pragma unroll
            for (int nt = 0; nt < 8; ++nt) {
                const int n = (ni * 8 + nt) * 16 + lane15;
                const float bb = base_b[n];
#pragma unroll
                for (int r = 0; r < 8; ++r) {
                    float v = acc[nt][r] + bb;
                    feat[(lmb + r) * 260 + n] = v > 0.f ? v : 0.f;
                }
            }
        }
        __syncthreads();

        if (j < 9) {
            const int mrow2 = chunk * 32 + m_ep;
            const float* fr = feat + m_ep * 260;
            const float4* rw = (const float4*)reg_w;   // [c][9] of float4
            float a0 = 0.f, a1 = 0.f, a2 = 0.f, a3 = 0.f, a4 = 0.f;
#pragma unroll 4
            for (int c = 0; c < 256; ++c) {
                float f = fr[c];
                float4 w4 = rw[c * 9 + j];
                float wc = cls_w[c * 9 + j];
                a0 += f * w4.x; a1 += f * w4.y;
                a2 += f * w4.z; a3 += f * w4.w;
                a4 += f * wc;
            }
            float4 rb = ((const float4*)reg_b)[j];
            a0 += rb.x; a1 += rb.y; a2 += rb.z; a3 += rb.w;
            float obj = 1.0f / (1.0f + expf(-(a4 + cls_b[j])));
            bool keep = (obj > 0.7f) & (a2 > 10.0f) & (a3 > 10.0f);
            size_t p = ((size_t)((bIdx * 64 + h0 + (mrow2 >> 6)) * 64 + (mrow2 & 63))) * 45;
            out[p + j * 4 + 0] = keep ? a0 : 0.f;
            out[p + j * 4 + 1] = keep ? a1 : 0.f;
            out[p + j * 4 + 2] = keep ? a2 : 0.f;
            out[p + j * 4 + 3] = keep ? a3 : 0.f;
            out[p + 36 + j]    = obj;
        }
    }
}

extern "C" void kernel_launch(void* const* d_in, const int* in_sizes, int n_in,
                              void* d_out, int out_size, void* d_ws, size_t ws_size,
                              hipStream_t stream) {
    (void)in_sizes; (void)n_in; (void)out_size; (void)ws_size;
    const float* x      = (const float*)d_in[0];
    const float* base_w = (const float*)d_in[1];
    const float* base_b = (const float*)d_in[2];
    const float* cls_w  = (const float*)d_in[3];
    const float* cls_b  = (const float*)d_in[4];
    const float* reg_w  = (const float*)d_in[5];
    const float* reg_b  = (const float*)d_in[6];
    _Float16* wp = (_Float16*)d_ws;        // 9*16*2*8192*2 B = 4.72 MB scratch

    rpn_pack_w<<<4608, 256, 0, stream>>>(base_w, wp);
    rpn_main<<<1024, 512, 0, stream>>>(x, base_b, cls_w, cls_b, reg_w, reg_b, wp,
                                       (float*)d_out);
}
